// VisionTransformer_1460288880802
// MI455X (gfx1250) — matrix operand order repaired and checked by reading, not compiled, not run on hardware
//
#include <hip/hip_runtime.h>
#include <hip/hip_bf16.h>

typedef __attribute__((ext_vector_type(16))) __bf16 v16bf;
typedef __attribute__((ext_vector_type(8)))  float  v8f;
typedef __attribute__((ext_vector_type(4)))  unsigned int u32x4;
typedef __attribute__((ext_vector_type(8)))  int i32x8;
typedef __attribute__((ext_vector_type(4)))  int i32x4;

#define TOK   151
#define DIMD  512
#define MROWS 2416   // 16*151

#if defined(__has_builtin)
#if __has_builtin(__builtin_amdgcn_tensor_load_to_lds) && __has_builtin(__builtin_amdgcn_s_wait_tensorcnt)
#define HAVE_TDM 1
#endif
#endif

// ---------------------------------------------------------------- cls + pos
__global__ void k_cls_pos(const float* __restrict__ cls, const float* __restrict__ pos,
                          float* __restrict__ t) {
  int b = blockIdx.x;
  for (int d = threadIdx.x; d < DIMD; d += 256)
    t[(size_t)(b * TOK) * DIMD + d] = cls[d] + pos[d];
}

// ------------------------------------------------- patch embed: [2400x27000]@[27000x512]
// grid (150, 4), 256 thr. Block owns 16 patch rows; wave w owns 16 output cols.
// K is walked one ii-slice (30x30 = 900 elems/patch) at a time. The TDM loads each
// patch's slice (2D tile 30x30, stride 150) into LDS *contiguously*, so the WMMA
// K index in LDS is linear -- no per-element 3D address math in the hot loop.
__global__ void k_patch_embed(const float* __restrict__ x, const float* __restrict__ ew,
                              const float* __restrict__ eb, const float* __restrict__ pos,
                              float* __restrict__ t) {
  __shared__ float As[16 * 900];   // 57.6 KB; only shared object -> LDS offset 0
  int tid = threadIdx.x, lane = tid & 31, wave = tid >> 5;
  int mblk = blockIdx.x * 16;
  int n0 = blockIdx.y * 128 + wave * 16;
  v8f c = {};
  int arow = lane & 15, koff = (lane < 16) ? 0 : 8;
  int bn = lane & 15, bk0 = (lane < 16) ? 0 : 8;

  for (int ii = 0; ii < 30; ++ii) {
#ifdef HAVE_TDM
    if (wave == 0 && lane == 0) {
      // one descriptor per patch row: 30x30 fp32 tile, tensor row stride 150
      for (int r = 0; r < 16; ++r) {
        int m = mblk + r;
        int b = m / 150, p = m % 150;
        int pa = p / 30, pb = (p / 5) % 6, pc = p % 5;
        unsigned long long gaddr = (unsigned long long)(uintptr_t)x +
            4ull * ((unsigned long long)b * 4050000ull +
                    (unsigned long long)((pa * 30 + ii) * 180 + pb * 30) * 150ull +
                    (unsigned long long)(pc * 30));
        u32x4 g0;
        g0[0] = 1u;                                   // count=1 (valid user D#)
        g0[1] = (unsigned)(r * 3600);                 // lds_addr (bytes)
        g0[2] = (unsigned)(gaddr & 0xFFFFFFFFu);      // global_addr[31:0]
        g0[3] = (unsigned)((gaddr >> 32) & 0x1FFFFFFu) | (2u << 30); // addr[56:32] | type=2
        i32x8 g1;
        g1[0] = (int)(2u << 16);                      // data_size = 4B
        g1[1] = (int)((150u & 0xFFFFu) << 16);        // tensor_dim0 lo16
        g1[2] = (int)((150u >> 16) | ((180u & 0xFFFFu) << 16)); // dim0 hi | dim1 lo
        g1[3] = (int)((180u >> 16) | (30u << 16));    // dim1 hi | tile_dim0=30
        g1[4] = (int)(30u);                           // tile_dim1=30, tile_dim2=0
        g1[5] = (int)150;                             // tensor_dim0_stride lo32 = 150
        g1[6] = 0;                                    // stride0 hi | stride1 lo (2D: unused)
        g1[7] = 0;
        i32x4 gz = {0, 0, 0, 0};
#if __clang_major__ >= 23
        i32x8 gz8 = {0, 0, 0, 0, 0, 0, 0, 0};
        __builtin_amdgcn_tensor_load_to_lds(g0, g1, gz, gz, gz8, 0);
#else
        __builtin_amdgcn_tensor_load_to_lds(g0, g1, gz, gz, 0);
#endif
      }
    }
    if (wave == 0) __builtin_amdgcn_s_wait_tensorcnt(0);
#else
    // fallback: cooperative gather of the 16x900 slice
    for (int i = tid; i < 16 * 900; i += 256) {
      int r = i / 900, kl = i % 900;
      int m = mblk + r;
      int b = m / 150, p = m % 150;
      int pa = p / 30, pb = (p / 5) % 6, pc = p % 5;
      int jj = kl / 30, kk = kl % 30;
      As[i] = x[(size_t)b * 4050000u +
                (size_t)((pa * 30 + ii) * 180 + (pb * 30 + jj)) * 150 + (pc * 30 + kk)];
    }
#endif
    __syncthreads();

    const float* pB = ew + (size_t)(n0 + bn) * 27000 + ii * 900;
    const float* pA = As + arow * 900;
    for (int kl = 0; kl < 900; kl += 32) {   // 29 steps, last has 4 valid (900=28*32+4)
      v16bf a, bb;
#pragma unroll
      for (int e = 0; e < 8; ++e) {
        int kloc = kl + koff + e;
        int kc2 = kloc < 900 ? kloc : 0;
        float va = pA[kc2];
        a[e] = (__bf16)(kloc < 900 ? va : 0.f);
      }
#pragma unroll
      for (int e = 0; e < 8; ++e) {
        int kloc = kl + 16 + koff + e;
        int kc2 = kloc < 900 ? kloc : 0;
        float va = pA[kc2];
        a[8 + e] = (__bf16)(kloc < 900 ? va : 0.f);
      }
#pragma unroll
      for (int e = 0; e < 16; ++e) {
        int kloc = kl + bk0 + e + (e & 8);
        int kc2 = kloc < 900 ? kloc : 0;
        float vb = pB[kc2];
        bb[e] = (__bf16)(kloc < 900 ? vb : 0.f);
      }
      c = __builtin_amdgcn_wmma_f32_16x16x32_bf16(false, a, false, bb, (short)0, c, false, false);
    }
    __syncthreads();
  }

  int cn = lane & 15, mbase = (lane < 16) ? 0 : 8;
#pragma unroll
  for (int v = 0; v < 8; ++v) {
    int m = mblk + mbase + v;
    int b = m / 150, p = m % 150, tok = 1 + p;
    int n = n0 + cn;
    t[((size_t)b * TOK + tok) * DIMD + n] = c[v] + eb[n] + pos[(size_t)tok * DIMD + n];
  }
}

// ---------------------------------------------------------------- layernorm (rows=grid.x)
__global__ void k_layernorm(const float* __restrict__ in, float* __restrict__ out,
                            const float* __restrict__ g, const float* __restrict__ bt) {
  int r = blockIdx.x, tid = threadIdx.x;
  const float* xr = in + (size_t)r * DIMD;
  float s = 0.f, s2 = 0.f;
  for (int d = tid; d < DIMD; d += 256) { float v = xr[d]; s += v; s2 += v * v; }
  __shared__ float rs[256], rs2[256];
  rs[tid] = s; rs2[tid] = s2; __syncthreads();
  for (int o = 128; o; o >>= 1) {
    if (tid < o) { rs[tid] += rs[tid + o]; rs2[tid] += rs2[tid + o]; }
    __syncthreads();
  }
  float mu = rs[0] / 512.f, var = rs2[0] / 512.f - mu * mu;
  float inv = rsqrtf(var + 1e-5f);
  float* orow = out + (size_t)r * DIMD;
  for (int d = tid; d < DIMD; d += 256) orow[d] = (xr[d] - mu) * inv * g[d] + bt[d];
}

// ------------------------------------- generic GEMM: C[m,n] = ep(A[m,:]@W[n,:] + bias[n])
// grid (ceil(M/64), N/64), 256 thr (8 waves = 4Mx2N, each wave 16x32)
__global__ void k_gemm(const float* __restrict__ A, const float* __restrict__ W,
                       const float* __restrict__ bias, const float* __restrict__ res,
                       float* __restrict__ C, int M, int N, int K, int act) {
  __shared__ float As[64][33];
  int tid = threadIdx.x, lane = tid & 31, wave = tid >> 5;
  int wm = wave >> 1, wn = wave & 1;
  int mblk = blockIdx.x * 64, nblk = blockIdx.y * 64;
  int m0 = mblk + wm * 16, n0 = nblk + wn * 32;
  v8f c0 = {}, c1 = {};
  int arow = lane & 15, koff = (lane < 16) ? 0 : 8;
  int bn = lane & 15, bk = (lane < 16) ? 0 : 8;
  for (int kc = 0; kc < K; kc += 32) {
    for (int i = tid; i < 64 * 32; i += 256) {
      int r = i >> 5, cc = i & 31;
      int m = mblk + r;
      As[r][cc] = (m < M) ? A[(size_t)m * K + kc + cc] : 0.f;
    }
    if (kc + 32 < K) {
      int pm = mblk + (tid & 63);
      if (pm < M) __builtin_prefetch(&A[(size_t)pm * K + kc + 32], 0, 0);
    }
    __syncthreads();
    v16bf a;
    const float* pA = &As[wm * 16 + arow][0];
#pragma unroll
    for (int e = 0; e < 8; ++e) a[e] = (__bf16)pA[koff + e];
#pragma unroll
    for (int e = 0; e < 8; ++e) a[8 + e] = (__bf16)pA[16 + koff + e];
    {
      const float* p = W + (size_t)(n0 + bn) * K + kc + bk;
      v16bf bb;
#pragma unroll
      for (int e = 0; e < 16; ++e) bb[e] = (__bf16)p[e + (e & 8)];
      c0 = __builtin_amdgcn_wmma_f32_16x16x32_bf16(false, a, false, bb, (short)0, c0, false, false);
    }
    {
      const float* p = W + (size_t)(n0 + 16 + bn) * K + kc + bk;
      v16bf bb;
#pragma unroll
      for (int e = 0; e < 16; ++e) bb[e] = (__bf16)p[e + (e & 8)];
      c1 = __builtin_amdgcn_wmma_f32_16x16x32_bf16(false, a, false, bb, (short)0, c1, false, false);
    }
    __syncthreads();
  }
  int cn = lane & 15, mbase = (lane < 16) ? 0 : 8;
#pragma unroll
  for (int v = 0; v < 8; ++v) {
    int m = m0 + mbase + v;
    if (m < M) {
#pragma unroll
      for (int half = 0; half < 2; ++half) {
        int n = n0 + half * 16 + cn;
        float val = half ? c1[v] : c0[v];
        if (bias) val += bias[n];
        if (act) val = 0.5f * val * (1.f + erff(val * 0.70710678f));
        if (res) val += res[(size_t)m * N + n];
        C[(size_t)m * N + n] = val;
      }
    }
  }
}

// -------------------------------------------- attention logits: per (b,head) 160x160 tile
__global__ void k_attn_logits(const float* __restrict__ qkv, float* __restrict__ lg) {
  int tid = threadIdx.x, lane = tid & 31, wave = tid >> 5;
  int b = blockIdx.x >> 3, hh = blockIdx.x & 7;
  const float* base = qkv + (size_t)b * TOK * 1536;
  float* L = lg + (size_t)blockIdx.x * 160 * 160;
  int arow = lane & 15, koff = (lane < 16) ? 0 : 8;
  int bn = lane & 15, bk0 = (lane < 16) ? 0 : 8;
  for (int tile = wave; tile < 100; tile += 8) {
    int mi = tile / 10, ni = tile % 10;
    int m0 = mi * 16, n0 = ni * 16;
    v8f c = {};
    int m = m0 + arow; if (m > 150) m = 150;
    int n = n0 + bn;   if (n > 150) n = 150;
    const float* qp = base + (size_t)m * 1536 + hh * 64;
    const float* kp = base + (size_t)n * 1536 + 512 + hh * 64;
#pragma unroll
    for (int kc = 0; kc < 64; kc += 32) {
      v16bf a, bb;
#pragma unroll
      for (int e = 0; e < 8; ++e) a[e] = (__bf16)qp[kc + koff + e];
#pragma unroll
      for (int e = 0; e < 8; ++e) a[8 + e] = (__bf16)qp[kc + 16 + koff + e];
#pragma unroll
      for (int e = 0; e < 16; ++e) bb[e] = (__bf16)kp[kc + bk0 + e + (e & 8)];
      c = __builtin_amdgcn_wmma_f32_16x16x32_bf16(false, a, false, bb, (short)0, c, false, false);
    }
    int cn = lane & 15, mbase = (lane < 16) ? 0 : 8;
#pragma unroll
    for (int v = 0; v < 8; ++v)
      L[(size_t)(m0 + mbase + v) * 160 + n0 + cn] = c[v] * 0.125f;
  }
}

// -------- row softmax -> bf16 w; also MAWS contribution (col-0 softmax over q) and w[0,:]
__global__ void k_attn_softmax(const float* __restrict__ lg, __bf16* __restrict__ wb,
                               float* __restrict__ ct, float* __restrict__ w0) {
  int tid = threadIdx.x, lane = tid & 31, wave = tid >> 5;
  const float* L = lg + (size_t)blockIdx.x * 160 * 160;
  __bf16* W = wb + (size_t)blockIdx.x * 160 * 160;
  for (int r = wave; r < 160; r += 8) {
    if (r < TOK) {
      const float* row = L + (size_t)r * 160;
      float mx = -1e30f;
      for (int n = lane; n < TOK; n += 32) mx = fmaxf(mx, row[n]);
      for (int off = 16; off; off >>= 1) mx = fmaxf(mx, __shfl_xor(mx, off));
      float s = 0.f;
      for (int n = lane; n < TOK; n += 32) s += __expf(row[n] - mx);
      for (int off = 16; off; off >>= 1) s += __shfl_xor(s, off);
      float inv = 1.f / s;
      for (int n = lane; n < 160; n += 32) {
        float v = (n < TOK) ? __expf(row[n] - mx) * inv : 0.f;
        W[(size_t)r * 160 + n] = (__bf16)v;
        if (r == 0 && n < TOK) w0[(size_t)blockIdx.x * 160 + n] = v;
      }
    } else {
      for (int n = lane; n < 160; n += 32) W[(size_t)r * 160 + n] = (__bf16)0.f;
    }
  }
  __shared__ float red[256];
  float v = (tid < TOK) ? L[(size_t)tid * 160] : -1e30f;
  red[tid] = v; __syncthreads();
  for (int o = 128; o; o >>= 1) { if (tid < o) red[tid] = fmaxf(red[tid], red[tid + o]); __syncthreads(); }
  float mx = red[0]; __syncthreads();
  float ev = (tid < TOK) ? __expf(v - mx) : 0.f;
  red[tid] = ev; __syncthreads();
  for (int o = 128; o; o >>= 1) { if (tid < o) red[tid] += red[tid + o]; __syncthreads(); }
  if (tid < TOK) ct[(size_t)blockIdx.x * 160 + tid] = ev / red[0];
}

// ---------------------------------------------------------------- o = w @ v  (per b,head)
__global__ void k_attn_av(const float* __restrict__ qkv, const __bf16* __restrict__ wb,
                          float* __restrict__ o) {
  int tid = threadIdx.x, lane = tid & 31, wave = tid >> 5;
  int b = blockIdx.x >> 3, hh = blockIdx.x & 7;
  const float* vbase = qkv + (size_t)b * TOK * 1536 + 1024 + hh * 64;
  const __bf16* W = wb + (size_t)blockIdx.x * 160 * 160;
  int arow = lane & 15, koff = (lane < 16) ? 0 : 8;
  int bn = lane & 15, bk0 = (lane < 16) ? 0 : 8;
  for (int tile = wave; tile < 40; tile += 8) {
    int mi = tile >> 2, ni = tile & 3;
    int m0 = mi * 16, n0 = ni * 16;
    v8f c = {};
    const __bf16* ap = W + (size_t)(m0 + arow) * 160;
#pragma unroll
    for (int kc = 0; kc < 160; kc += 32) {
      v16bf a, bb;
#pragma unroll
      for (int e = 0; e < 8; ++e) a[e] = ap[kc + koff + e];
#pragma unroll
      for (int e = 0; e < 8; ++e) a[8 + e] = ap[kc + 16 + koff + e];
#pragma unroll
      for (int e = 0; e < 16; ++e) {
        int kd = kc + bk0 + e + (e & 8); if (kd > 150) kd = 150;
        bb[e] = (__bf16)vbase[(size_t)kd * 1536 + n0 + bn];
      }
      c = __builtin_amdgcn_wmma_f32_16x16x32_bf16(false, a, false, bb, (short)0, c, false, false);
    }
    int cn = lane & 15, mbase = (lane < 16) ? 0 : 8;
#pragma unroll
    for (int v = 0; v < 8; ++v) {
      int m = m0 + mbase + v;
      if (m < TOK) o[((size_t)b * TOK + m) * DIMD + hh * 64 + n0 + cn] = c[v];
    }
  }
}

// ------------------------------------- MAWS top-6, mean pool, fc_norm, linear head
__global__ void k_head(const float* __restrict__ t, const float* __restrict__ ct,
                       const float* __restrict__ w0, const float* __restrict__ fw,
                       const float* __restrict__ fb, const float* __restrict__ hw,
                       const float* __restrict__ hb, float* __restrict__ out) {
  int b = blockIdx.x, tid = threadIdx.x;
  __shared__ float sc[160];
  __shared__ int idx[6];
  __shared__ float pooled[512];
  __shared__ float rs[256], rs2[256];
  for (int n = tid; n < 160; n += 256) {
    if (n < TOK) {
      float cm = 0.f, wm = 0.f;
      for (int hh = 0; hh < 8; ++hh) { cm += ct[(size_t)(b * 8 + hh) * 160 + n]; wm += w0[(size_t)(b * 8 + hh) * 160 + n]; }
      sc[n] = (cm * 0.125f) * (wm * 0.125f);
    } else sc[n] = -1e30f;
  }
  __syncthreads();
  if (tid == 0) {
    for (int s = 0; s < 6; ++s) {
      int best = 0; float bv = -1e30f;
      for (int n = 0; n < TOK; ++n) if (sc[n] > bv) { bv = sc[n]; best = n; }
      idx[s] = best; sc[best] = -1e30f;
    }
  }
  __syncthreads();
  for (int d = tid; d < DIMD; d += 256) {
    float s = 0.f;
    for (int n = 1; n < TOK; ++n) s += t[((size_t)b * TOK + n) * DIMD + d];
    for (int s6 = 0; s6 < 6; ++s6) s += t[((size_t)b * TOK + idx[s6]) * DIMD + d];
    pooled[d] = s * (1.f / 156.f);
  }
  __syncthreads();
  float s = 0.f, s2 = 0.f;
  for (int d = tid; d < DIMD; d += 256) { float v = pooled[d]; s += v; s2 += v * v; }
  rs[tid] = s; rs2[tid] = s2; __syncthreads();
  for (int o = 128; o; o >>= 1) { if (tid < o) { rs[tid] += rs[tid + o]; rs2[tid] += rs2[tid + o]; } __syncthreads(); }
  float mu = rs[0] / 512.f, var = rs2[0] / 512.f - mu * mu, inv = rsqrtf(var + 1e-5f);
  __syncthreads();
  float p0 = 0.f, p1 = 0.f;
  for (int d = tid; d < DIMD; d += 256) {
    float v = (pooled[d] - mu) * inv * fw[d] + fb[d];
    p0 += v * hw[d]; p1 += v * hw[DIMD + d];
  }
  rs[tid] = p0; rs2[tid] = p1; __syncthreads();
  for (int o = 128; o; o >>= 1) { if (tid < o) { rs[tid] += rs[tid + o]; rs2[tid] += rs2[tid + o]; } __syncthreads(); }
  if (tid == 0) { out[b * 2 + 0] = rs[0] + hb[0]; out[b * 2 + 1] = rs2[0] + hb[1]; }
}

extern "C" void kernel_launch(void* const* d_in, const int* in_sizes, int n_in,
                              void* d_out, int out_size, void* d_ws, size_t ws_size,
                              hipStream_t stream) {
  (void)in_sizes; (void)n_in; (void)out_size; (void)ws_size;
  const float* x     = (const float*)d_in[0];
  const float* ew    = (const float*)d_in[1];
  const float* eb    = (const float*)d_in[2];
  const float* cls   = (const float*)d_in[3];
  const float* pos   = (const float*)d_in[4];
  const float* ln1w  = (const float*)d_in[5];
  const float* ln1b  = (const float*)d_in[6];
  const float* qkvw  = (const float*)d_in[7];
  const float* projw = (const float*)d_in[8];
  const float* projb = (const float*)d_in[9];
  const float* ln2w  = (const float*)d_in[10];
  const float* ln2b  = (const float*)d_in[11];
  const float* fc1w  = (const float*)d_in[12];
  const float* fc1b  = (const float*)d_in[13];
  const float* fc2w  = (const float*)d_in[14];
  const float* fc2b  = (const float*)d_in[15];
  const float* fcnw  = (const float*)d_in[16];
  const float* fcnb  = (const float*)d_in[17];
  const float* headw = (const float*)d_in[18];
  const float* headb = (const float*)d_in[19];

  float* ws  = (float*)d_ws;
  float* t   = ws;                         // 2416*512
  float* h   = ws + 1236992;               // 2416*512
  float* qkv = ws + 2473984;               // 2416*1536
  float* ob  = ws + 6184960;               // 2416*512
  float* u   = ws + 7421952;               // 2416*2048
  float* lg  = ws + 12369920;              // 128*160*160 f32 logits
  __bf16* wb = (__bf16*)(ws + 15646720);   // 128*160*160 bf16 weights
  float* ct  = ws + 17285120;              // 16*8*160 contribution
  float* w0  = ws + 17305600;              // 16*8*160 w[0,:]

  k_cls_pos<<<16, 256, 0, stream>>>(cls, pos, t);
  k_patch_embed<<<dim3(150, 4), 256, 0, stream>>>(x, ew, eb, pos, t);
  for (int i = 0; i < 8; ++i) {
    k_layernorm<<<MROWS, 256, 0, stream>>>(t, h, ln1w + i * 512, ln1b + i * 512);
    k_gemm<<<dim3(38, 24), 256, 0, stream>>>(h, qkvw + (size_t)i * 1536 * 512, nullptr, nullptr,
                                             qkv, MROWS, 1536, 512, 0);
    k_attn_logits<<<128, 256, 0, stream>>>(qkv, lg);
    k_attn_softmax<<<128, 256, 0, stream>>>(lg, wb, ct, w0);
    k_attn_av<<<128, 256, 0, stream>>>(qkv, wb, ob);
    k_gemm<<<dim3(38, 8), 256, 0, stream>>>(ob, projw + (size_t)i * 512 * 512, projb + i * 512,
                                            t, t, MROWS, 512, 512, 0);
    k_layernorm<<<MROWS, 256, 0, stream>>>(t, h, ln2w + i * 512, ln2b + i * 512);
    k_gemm<<<dim3(38, 32), 256, 0, stream>>>(h, fc1w + (size_t)i * 2048 * 512, fc1b + i * 2048,
                                             nullptr, u, MROWS, 2048, 512, 1);
    k_gemm<<<dim3(38, 8), 256, 0, stream>>>(u, fc2w + (size_t)i * 512 * 2048, fc2b + i * 512,
                                            t, t, MROWS, 512, 2048, 0);
  }
  k_head<<<16, 256, 0, stream>>>(t, ct, w0, fcnw, fcnb, headw, headb, (float*)d_out);
}
